// PICL_23596550324769
// MI455X (gfx1250) — matrix operand order repaired and checked by reading, not compiled, not run on hardware
//
#include <hip/hip_runtime.h>
#include <hip/hip_bf16.h>
#include <stdint.h>

typedef __attribute__((ext_vector_type(16))) __bf16 v16bf;
typedef __attribute__((ext_vector_type(8)))  float  v8f;
typedef int v4i __attribute__((vector_size(16)));   // matches builtin's V4i param

#define BATCH 4096
#define FEA   256
#define CLSN  100
#define BM 128
#define BN 128
#define BK 32
#define LDP (BK + 8)   // padded LDS row stride in bf16 elements (80B, 16B-aligned)

#if defined(__gfx1250__) && __has_builtin(__builtin_amdgcn_global_load_async_to_lds_b128) && __has_builtin(__builtin_amdgcn_s_wait_asynccnt)
#define USE_ASYNC 1
#else
#define USE_ASYNC 0
#endif

// generic -> addrspace casts (LDS generic addresses keep the LDS offset in low 32 bits)
#define AS3(p) ((__attribute__((address_space(3))) v4i*)(unsigned int)(uintptr_t)(p))
#define AS1(p) ((__attribute__((address_space(1))) v4i*)(uintptr_t)(p))

__device__ __forceinline__ unsigned short f2bf(float f) {
  unsigned int u = __float_as_uint(f);
  unsigned int r = u + 0x7FFFu + ((u >> 16) & 1u);   // round-to-nearest-even
  return (unsigned short)(r >> 16);
}

// ---------------- fp32 -> bf16 convert ----------------
__global__ void cvt_bf16_kernel(const float* __restrict__ in,
                                unsigned short* __restrict__ out, int n) {
  int i = blockIdx.x * blockDim.x + threadIdx.x;
  if (i < n) out[i] = f2bf(in[i]);
}

// ---------------- W (f32, KxN) -> Wt (bf16, NxK) tiled transpose ----------------
__global__ __launch_bounds__(256) void transpose_cvt_kernel(
    const float* __restrict__ W, unsigned short* __restrict__ Wt, int K, int N)
{
  __shared__ float tile[32][33];
  int n0 = blockIdx.x * 32, k0 = blockIdx.y * 32;
  int tx = threadIdx.x & 31, ty = threadIdx.x >> 5;   // 32x8
  #pragma unroll
  for (int j = 0; j < 4; ++j)
    tile[ty + 8 * j][tx] = W[(size_t)(k0 + ty + 8 * j) * N + n0 + tx];
  __syncthreads();
  #pragma unroll
  for (int j = 0; j < 4; ++j)
    Wt[(size_t)(n0 + ty + 8 * j) * K + k0 + tx] = f2bf(tile[tx][ty + 8 * j]);
}

// ---------------- GEMM: C(f32, MxN) = A(bf16, MxK) @ Bt(bf16, NxK)^T + bias ----------------
// Double-buffered LDS; tiles staged with GLOBAL_LOAD_ASYNC_TO_LDS_B128 when available.
__global__ __launch_bounds__(256) void gemm_bf16_wmma(
    const unsigned short* __restrict__ A, const unsigned short* __restrict__ Bt,
    const float* __restrict__ bias, float* __restrict__ C,
    int M, int N, int K)
{
  __shared__ unsigned short As[2][BM * LDP];   // A tile [m][k]
  __shared__ unsigned short Bs[2][BN * LDP];   // B tile [n][k]

  const int tid  = threadIdx.x;
  const int lane = tid & 31;
  const int wid  = tid >> 5;
  const int wm   = (wid & 3) * 32;   // 4 waves along M
  const int wn   = (wid >> 2) * 64;  // 2 waves along N
  const int l15  = lane & 15;
  const int hi   = lane >> 4;
  const int bm   = blockIdx.y * BM;
  const int bn   = blockIdx.x * BN;

  const v8f vzero = {0.f,0.f,0.f,0.f,0.f,0.f,0.f,0.f};
  v8f acc[2][4];
  #pragma unroll
  for (int r = 0; r < 2; ++r)
    #pragma unroll
    for (int c = 0; c < 4; ++c) acc[r][c] = vzero;

  // staging coordinates: each thread copies one 32B row-segment of A and of Bt
  const int srow = tid >> 1;            // 0..127
  const int sseg = (tid & 1) * 16;      // element offset 0 or 16

  const unsigned short* gA = A  + (size_t)(bm + srow) * K + sseg;
  const unsigned short* gB = Bt + (size_t)(bn + srow) * K + sseg;
  const int lofs = srow * LDP + sseg;

  auto stage = [&](int buf, int k0) {
#if USE_ASYNC
    __builtin_amdgcn_global_load_async_to_lds_b128(AS1(gA + k0), AS3(&As[buf][lofs]), 0, 0);
    __builtin_amdgcn_global_load_async_to_lds_b128(AS1(gA + k0), AS3(&As[buf][lofs]), 16, 0);
    __builtin_amdgcn_global_load_async_to_lds_b128(AS1(gB + k0), AS3(&Bs[buf][lofs]), 0, 0);
    __builtin_amdgcn_global_load_async_to_lds_b128(AS1(gB + k0), AS3(&Bs[buf][lofs]), 16, 0);
#else
    const uint4* ga4 = (const uint4*)(gA + k0);
    const uint4* gb4 = (const uint4*)(gB + k0);
    uint4* la = (uint4*)(&As[buf][lofs]);
    uint4* lb = (uint4*)(&Bs[buf][lofs]);
    la[0] = ga4[0]; la[1] = ga4[1];
    lb[0] = gb4[0]; lb[1] = gb4[1];
    __builtin_prefetch(gA + k0 + BK, 0, 0);
    __builtin_prefetch(gB + k0 + BK, 0, 0);
#endif
  };

  const int nk = K / BK;
  stage(0, 0);

  for (int t = 0; t < nk; ++t) {
    const int cur = t & 1;
    const bool more = (t + 1 < nk);
    if (more) stage(cur ^ 1, (t + 1) * BK);
#if USE_ASYNC
    if (more) __builtin_amdgcn_s_wait_asynccnt(4);   // tile t's 4 loads done
    else      __builtin_amdgcn_s_wait_asynccnt(0);
#endif
    __syncthreads();

    // ---- load fragments (ISA 7.12.2 16-bit layouts)
    v16bf afrag[2];
    #pragma unroll
    for (int r = 0; r < 2; ++r) {
      int m = wm + r * 16 + l15;
      uint4* d = (uint4*)&afrag[r];
      d[0] = *(const uint4*)&As[cur][m * LDP + hi * 8];       // K = hi*8 .. +7
      d[1] = *(const uint4*)&As[cur][m * LDP + 16 + hi * 8];  // K = 16+hi*8 .. +7
    }
    v16bf bfrag[4];
    #pragma unroll
    for (int c = 0; c < 4; ++c) {
      int n = wn + c * 16 + l15;
      uint4* d = (uint4*)&bfrag[c];
      d[0] = *(const uint4*)&Bs[cur][n * LDP + hi * 8];       // K = hi*8 .. +7
      d[1] = *(const uint4*)&Bs[cur][n * LDP + 16 + hi * 8];  // K = 16+hi*8 .. +7
    }

    // ---- 2x4 grid of v_wmma_f32_16x16x32_bf16
    #pragma unroll
    for (int r = 0; r < 2; ++r)
      #pragma unroll
      for (int c = 0; c < 4; ++c)
        acc[r][c] = __builtin_amdgcn_wmma_f32_16x16x32_bf16(
            false, afrag[r], false, bfrag[c], (short)0, acc[r][c], false, false);

    __syncthreads();   // protect buf[cur] from next-next stage overwrite
  }

  // ---- epilogue: bias add + store (C/D layout: VGPR v -> M = v + hi*8, N = l15)
  #pragma unroll
  for (int r = 0; r < 2; ++r) {
    #pragma unroll
    for (int c = 0; c < 4; ++c) {
      int n = bn + wn + c * 16 + l15;
      float bv = bias[n];
      int mbase = bm + wm + r * 16 + hi * 8;
      #pragma unroll
      for (int v = 0; v < 8; ++v)
        C[(size_t)(mbase + v) * N + n] = acc[r][c][v] + bv;
    }
  }
}

// ---------------- BN batch stats -> folded scale/shift ----------------
__global__ void bn_stats_kernel(const float* __restrict__ H,
                                const float* __restrict__ g,
                                const float* __restrict__ beta,
                                float* __restrict__ scale, float* __restrict__ shift,
                                int M, int N)
{
  int col = blockIdx.x * blockDim.x + threadIdx.x;
  if (col >= N) return;
  float s = 0.f, s2 = 0.f;
  for (int m = 0; m < M; ++m) {
    float v = H[(size_t)m * N + col];
    s += v; s2 += v * v;
  }
  float inv = 1.f / (float)M;
  float mu = s * inv;
  float var = fmaxf(s2 * inv - mu * mu, 0.f);
  float sc = g[col] * rsqrtf(var + 1e-5f);
  scale[col] = sc;
  shift[col] = beta[col] - mu * sc;
}

// ---------------- fused BN + ReLU ----------------
__global__ void bn_relu_bf16_kernel(const float* __restrict__ H,
                                    const float* __restrict__ scale,
                                    const float* __restrict__ shift,
                                    unsigned short* __restrict__ out, int N, int total)
{
  int i = blockIdx.x * blockDim.x + threadIdx.x;
  if (i >= total) return;
  int n = i % N;
  out[i] = f2bf(fmaxf(0.f, H[i] * scale[n] + shift[n]));
}

__global__ void bn_relu_f32_kernel(const float* __restrict__ H,
                                   const float* __restrict__ scale,
                                   const float* __restrict__ shift,
                                   float* __restrict__ out, int N, int total)
{
  int i = blockIdx.x * blockDim.x + threadIdx.x;
  if (i >= total) return;
  int n = i % N;
  out[i] = fmaxf(0.f, H[i] * scale[n] + shift[n]);
}

// ---------------- classifier: logits = q @ Wc + bc ----------------
__global__ void classifier_kernel(const float* __restrict__ q,
                                  const float* __restrict__ Wc,
                                  const float* __restrict__ bc,
                                  float* __restrict__ out, int M, int Cc, int Kf)
{
  int i = blockIdx.x * blockDim.x + threadIdx.x;
  if (i >= M * Cc) return;
  int m = i / Cc, c = i % Cc;
  float acc = bc[c];
  const float* qr = q + (size_t)m * Kf;
  for (int k = 0; k < Kf; ++k) acc += qr[k] * Wc[k * Cc + c];
  out[i] = acc;
}

// ---------------- prototype EMA weights (backward label scan) ----------------
__global__ void proto_w_kernel(const int* __restrict__ labels,
                               float* __restrict__ w, float* __restrict__ countpow,
                               int B, float mm)
{
  int c = threadIdx.x;
  if (c >= CLSN) return;
  int cnt = 0;
  for (int i = B - 1; i >= 0; --i) {
    if (labels[i] == c) {
      w[i] = (1.f - mm) * __powf(mm, (float)cnt);
      ++cnt;
    }
  }
  countpow[c] = __powf(mm, (float)cnt);
}

// ---------------- prototype accumulate + L2 normalize ----------------
__global__ __launch_bounds__(256) void proto_update_kernel(
    const float* __restrict__ proto0, const float* __restrict__ k,
    const int* __restrict__ labels, const float* __restrict__ w,
    const float* __restrict__ countpow, float* __restrict__ out, int B)
{
  __shared__ float red[256];
  int c = blockIdx.x;
  int f = threadIdx.x;
  float acc = proto0[c * FEA + f] * countpow[c];
  for (int i = 0; i < B; ++i)
    if (labels[i] == c) acc += w[i] * k[(size_t)i * FEA + f];
  red[f] = acc * acc;
  __syncthreads();
  for (int s = 128; s > 0; s >>= 1) {
    if (f < s) red[f] += red[f + s];
    __syncthreads();
  }
  float nrm = fmaxf(sqrtf(red[0]), 1e-12f);
  out[c * FEA + f] = acc / nrm;
}

extern "C" void kernel_launch(void* const* d_in, const int* in_sizes, int n_in,
                              void* d_out, int out_size, void* d_ws, size_t ws_size,
                              hipStream_t stream)
{
  (void)in_sizes; (void)n_in; (void)out_size; (void)ws_size;

  // jax sorted-dict flatten order: X0, X1, labels, params{cls,enc,pi}, protos
  const float* X0     = (const float*)d_in[0];
  const float* X1     = (const float*)d_in[1];
  const int*   labels = (const int*)d_in[2];
  const int I_CLS0_W = 3, I_CLS0_B = 4, I_CLS1_W = 5, I_CLS1_B = 6;
  const int I_ENC0 = 7, I_ENC1 = 23, I_PI0 = 39, I_PI1 = 55;
  const int I_P0 = 71, I_P1 = 72;

  float* out    = (float*)d_out;
  float* fea0   = out;
  float* fea1   = out + (size_t)8192 * 256;
  float* logit0 = out + (size_t)2 * 8192 * 256;
  float* logit1 = logit0 + (size_t)4096 * 100;
  float* pout0  = logit1 + (size_t)4096 * 100;
  float* pout1  = pout0 + (size_t)100 * 256;

  // workspace carve (256B aligned)
  char* p = (char*)d_ws;
  auto carve = [&](size_t bytes) -> void* {
    void* r = (void*)p;
    p += (bytes + 255) & ~(size_t)255;
    return r;
  };
  unsigned short* bfX0 = (unsigned short*)carve((size_t)4096 * 2048 * 2);
  unsigned short* bfX1 = (unsigned short*)carve((size_t)4096 * 1024 * 2);
  float*          H    = (float*)carve((size_t)4096 * 2048 * 4);
  unsigned short* act  = (unsigned short*)carve((size_t)4096 * 2048 * 2);
  unsigned short* Wt   = (unsigned short*)carve((size_t)2048 * 2048 * 2);
  float*          scale    = (float*)carve(2048 * 4);
  float*          shift    = (float*)carve(2048 * 4);
  float*          w        = (float*)carve(4096 * 4);
  float*          countpow = (float*)carve(128 * 4);

  // inputs to bf16 (each reused by two encoder chains)
  { int n = 4096 * 2048; cvt_bf16_kernel<<<(n + 255) / 256, 256, 0, stream>>>(X0, bfX0, n); }
  { int n = 4096 * 1024; cvt_bf16_kernel<<<(n + 255) / 256, 256, 0, stream>>>(X1, bfX1, n); }

  auto run_encoder = [&](const unsigned short* bf_in, int Kin, int base, float* dest) {
    int dims[5] = {Kin, 2048, 2048, 1024, 256};
    const unsigned short* ain = bf_in;
    for (int l = 0; l < 4; ++l) {
      const float* Wl = (const float*)d_in[base + l * 4 + 0];
      const float* bl = (const float*)d_in[base + l * 4 + 1];
      const float* gl = (const float*)d_in[base + l * 4 + 2];
      const float* bb = (const float*)d_in[base + l * 4 + 3];
      int Kl = dims[l], Nl = dims[l + 1];
      // W (Kl x Nl f32) -> Wt (Nl x Kl bf16)
      transpose_cvt_kernel<<<dim3(Nl / 32, Kl / 32), 256, 0, stream>>>(Wl, Wt, Kl, Nl);
      dim3 grid(Nl / BN, 4096 / BM);
      gemm_bf16_wmma<<<grid, 256, 0, stream>>>(ain, Wt, bl, H, 4096, Nl, Kl);
      bn_stats_kernel<<<(Nl + 255) / 256, 256, 0, stream>>>(H, gl, bb, scale, shift, 4096, Nl);
      int tot = 4096 * Nl;
      if (l < 3) {
        bn_relu_bf16_kernel<<<(tot + 255) / 256, 256, 0, stream>>>(H, scale, shift, act, Nl, tot);
        ain = act;
      } else {
        bn_relu_f32_kernel<<<(tot + 255) / 256, 256, 0, stream>>>(H, scale, shift, dest, Nl, tot);
      }
    }
  };

  run_encoder(bfX0, 2048, I_ENC0, fea0);                       // q0
  run_encoder(bfX1, 1024, I_PI0,  fea0 + (size_t)4096 * 256);  // k0
  run_encoder(bfX1, 1024, I_ENC1, fea1);                       // q1
  run_encoder(bfX0, 2048, I_PI1,  fea1 + (size_t)4096 * 256);  // k1

  {
    int tot = 4096 * 100;
    classifier_kernel<<<(tot + 255) / 256, 256, 0, stream>>>(
        fea0, (const float*)d_in[I_CLS0_W], (const float*)d_in[I_CLS0_B], logit0, 4096, 100, 256);
    classifier_kernel<<<(tot + 255) / 256, 256, 0, stream>>>(
        fea1, (const float*)d_in[I_CLS1_W], (const float*)d_in[I_CLS1_B], logit1, 4096, 100, 256);
  }

  proto_w_kernel<<<1, 128, 0, stream>>>(labels, w, countpow, 4096, 0.99f);
  proto_update_kernel<<<CLSN, 256, 0, stream>>>(
      (const float*)d_in[I_P0], fea0 + (size_t)4096 * 256, labels, w, countpow, pout0, 4096);
  proto_update_kernel<<<CLSN, 256, 0, stream>>>(
      (const float*)d_in[I_P1], fea1 + (size_t)4096 * 256, labels, w, countpow, pout1, 4096);
}
